// ScOTLayer_17789754540129
// MI455X (gfx1250) — matrix operand order repaired and checked by reading, not compiled, not run on hardware
//
#include <hip/hip_runtime.h>
#include <cstdint>

typedef __attribute__((ext_vector_type(16))) __bf16 v16bf;
typedef __attribute__((ext_vector_type(8)))  float  v8f;

#define DI __device__ __forceinline__

// ---------------- problem dims ----------------
constexpr int Bn  = 8, Hh = 112, Wd = 112, Cc = 192, NHn = 6, HDm = 32;
constexpr int WSw = 7, SSs = 3, Ntok = 49, NWn = (Hh / WSw) * (Wd / WSw); // 256
constexpr int Mrows = Bn * Hh * Wd;                                       // 100352
constexpr float EPSc    = 1e-5f;
constexpr float LOG100c = 4.6051701859880914f;

DI float  bf2f(__bf16 x) { return (float)x; }
DI __bf16 f2bf(float x)  { return (__bf16)x; }

DI v8f wmma_bf16(v16bf a, v16bf b, v8f c) {
  // D = A(16x32 bf16) x B(32x16 bf16) + C(16x16 f32)
  return __builtin_amdgcn_wmma_f32_16x16x32_bf16(false, a, false, b, (short)0, c,
                                                 false, false);
}

// A-fragment K index: lanes0-15 -> K 0-7,16-23 ; lanes16-31 -> K 8-15,24-31
DI int a_kidx(int e, int hi) { return e + ((e & 8) ? 8 : 0) + (hi ? 8 : 0); }
// B-fragment K index: lanes0-15 -> K 0-7,16-23 ; lanes16-31 -> K 8-15,24-31
DI int b_kidx(int e, int hi) { return e + ((e & 8) ? 8 : 0) + (hi ? 8 : 0); }

// CDNA5 async copy: 16 bytes per lane, global -> LDS, tracked by ASYNCcnt
DI void async_copy16(uint32_t lds_addr, const void* gaddr) {
  asm volatile("global_load_async_to_lds_b128 %0, %1, off"
               :: "v"(lds_addr), "v"((unsigned long long)(uintptr_t)gaddr)
               : "memory");
}
DI void wait_async0() { asm volatile("s_wait_asynccnt 0" ::: "memory"); }

// ---------------- CPB table (pre-sigmoided) + logit scales ----------------
__global__ void cpb_kernel(const float* __restrict__ logit_scale,
                           const float* __restrict__ w0, const float* __restrict__ b0,
                           const float* __restrict__ w1, float* __restrict__ bias_t,
                           float* __restrict__ scales) {
  const int tid = threadIdx.x;
  if (tid < 169) {
    float ry = (float)(tid / 13 - 6);
    float rx = (float)(tid % 13 - 6);
    auto cvt = [](float r) {
      float t = r * (8.f / 6.f);
      float s = (t > 0.f) ? 1.f : ((t < 0.f) ? -1.f : 0.f);
      return s * log2f(fabsf(t) + 1.f) * (1.f / log2f(8.f));
    };
    float t0 = cvt(ry), t1 = cvt(rx);
    float acc[NHn] = {};
    for (int j = 0; j < 512; ++j) {
      float h = fmaxf(t0 * w0[j] + t1 * w0[512 + j] + b0[j], 0.f);
#pragma unroll
      for (int n = 0; n < NHn; ++n) acc[n] += h * w1[j * NHn + n];
    }
#pragma unroll
    for (int n = 0; n < NHn; ++n)   // fold 16*sigmoid here, once
      bias_t[tid * NHn + n] = 16.f / (1.f + __expf(-acc[n]));
  }
  if (tid >= 192 && tid < 192 + NHn) {
    int h = tid - 192;
    scales[h] = __expf(fminf(logit_scale[h], LOG100c));
  }
}

// ---------------- generic tiled WMMA GEMM: C = A @ W + bias ----------------
enum { EPI_F32 = 0, EPI_BF16 = 1, EPI_GELU_BF16 = 2 };

template <typename AT, int EPI>
__global__ __launch_bounds__(256) void gemm_ws(const AT* __restrict__ A,
                                               const float* __restrict__ W,
                                               const float* __restrict__ bias,
                                               void* __restrict__ Cout, int K, int N) {
  __shared__ __bf16 As[64][40];  // [m][k]
  __shared__ __bf16 Bs[64][40];  // [n][k]
  const int tid = threadIdx.x;
  const int bm = blockIdx.y * 64, bn = blockIdx.x * 64;
  const int wv = tid >> 5, lane = tid & 31;
  const int wm = wv & 3, wn = wv >> 2;
  const int mrow = lane & 15, hi = lane >> 4;

  v8f acc[2] = {};
  for (int k0 = 0; k0 < K; k0 += 32) {
#pragma unroll
    for (int p = 0; p < 8; ++p) {
      int i = tid + p * 256;
      int r = i >> 5, c = i & 31;
      As[r][c] = f2bf((float)A[(long)(bm + r) * K + k0 + c]);
      int nn = i & 63, kk = i >> 6;
      Bs[nn][kk] = f2bf(W[(long)(k0 + kk) * N + bn + nn]);
    }
    if (k0 + 32 < K) {  // CDNA5 global_prefetch of the next k-tile
      __builtin_prefetch(&A[(long)(bm + (tid >> 2)) * K + k0 + 32], 0, 1);
      __builtin_prefetch(&W[(long)(k0 + 32 + (tid >> 5) * 4) * N + bn], 0, 1);
    }
    __syncthreads();
    v16bf a;
#pragma unroll
    for (int e = 0; e < 16; ++e) a[e] = As[wm * 16 + mrow][a_kidx(e, hi)];
    v16bf b0, b1;
#pragma unroll
    for (int e = 0; e < 16; ++e) {
      int kk = b_kidx(e, hi);
      b0[e] = Bs[wn * 32 + mrow][kk];
      b1[e] = Bs[wn * 32 + 16 + mrow][kk];
    }
    acc[0] = wmma_bf16(a, b0, acc[0]);
    acc[1] = wmma_bf16(a, b1, acc[1]);
    __syncthreads();
  }
#pragma unroll
  for (int s = 0; s < 2; ++s)
#pragma unroll
    for (int r = 0; r < 8; ++r) {
      int m = wm * 16 + hi * 8 + r;
      int n = wn * 32 + s * 16 + mrow;
      long idx = (long)(bm + m) * N + (bn + n);
      float v = acc[s][r] + (bias ? bias[bn + n] : 0.f);
      if constexpr (EPI == EPI_F32) {
        ((float*)Cout)[idx] = v;
      } else if constexpr (EPI == EPI_BF16) {
        ((__bf16*)Cout)[idx] = f2bf(v);
      } else {
        float g = 0.5f * v * (1.f + tanhf(0.7978845608028654f * (v + 0.044715f * v * v * v)));
        ((__bf16*)Cout)[idx] = f2bf(g);
      }
    }
}

// ---------------- cosine-attn normalization (in place, scale folded into Q) ----------------
__global__ __launch_bounds__(256) void norm_qk_kernel(__bf16* __restrict__ Qn,
                                                      __bf16* __restrict__ Kn,
                                                      const float* __restrict__ scales) {
  int idx = blockIdx.x * 256 + threadIdx.x;
  if (idx >= Mrows * NHn) return;
  int row = idx / NHn, h = idx % NHn;
  __bf16* q = Qn + (long)row * Cc + h * HDm;
  __bf16* k = Kn + (long)row * Cc + h * HDm;
  float sq = 0.f, sk = 0.f;
#pragma unroll
  for (int d = 0; d < HDm; ++d) {
    float a = bf2f(q[d]); sq += a * a;
    float b = bf2f(k[d]); sk += b * b;
  }
  float iq = scales[h] / fmaxf(sqrtf(sq), 1e-12f);
  float ik = 1.f / fmaxf(sqrtf(sk), 1e-12f);
#pragma unroll
  for (int d = 0; d < HDm; ++d) {
    q[d] = f2bf(bf2f(q[d]) * iq);
    k[d] = f2bf(bf2f(k[d]) * ik);
  }
}

// ---------------- shifted-window attention: one block per window, wave = head ----------------
constexpr int TSTR = 200;  // q/k/v LDS row stride (bf16): 400B rows, 16B aligned, bank-rotated
constexpr int SSTR = 66;   // S row stride (f32)  -> conflict-free softmax columns
constexpr int PSTR = 66;   // P row stride (bf16)
constexpr int ATTN_SMEM =
    64 * 4 + 64 * 4 + 1024 * 4 + 3 * 64 * TSTR * 2 + 6 * 64 * SSTR * 4 + 6 * 64 * PSTR * 2;

__global__ __launch_bounds__(192) void attn_kernel(const __bf16* __restrict__ Qn,
                                                   const __bf16* __restrict__ Kn,
                                                   const __bf16* __restrict__ Vb,
                                                   const float* __restrict__ bias_t,
                                                   float* __restrict__ O) {
  extern __shared__ char smraw[];
  int*    rowoff = (int*)smraw;             // [64]
  int*    rid    = rowoff + 64;             // [64] shift-mask region ids
  float*  biasS  = (float*)(rid + 64);      // [1024] staged 16*sigmoid(cpb) table
  __bf16* qs     = (__bf16*)(biasS + 1024); // [64][TSTR]
  __bf16* ks     = qs + 64 * TSTR;
  __bf16* vs     = ks + 64 * TSTR;
  float*  Sb     = (float*)(vs + 64 * TSTR);   // 6 x [64][SSTR]
  __bf16* Pb     = (__bf16*)(Sb + 6 * 64 * SSTR);

  const int tid  = threadIdx.x;
  const int wblk = blockIdx.x;
  const int bimg = wblk / NWn;
  const int wid  = wblk % NWn;
  const int wy = wid / (Wd / WSw), wx = wid % (Wd / WSw);

  for (int i = tid; i < 169 * NHn; i += 192) biasS[i] = bias_t[i];
  if (tid < 64) {
    int t = tid, ro = 0, id = 0;
    if (t < Ntok) {
      int ty = t / WSw, tx = t % WSw;
      int hs = wy * WSw + ty, wsp = wx * WSw + tx;      // shifted-image coords
      int oh = (hs + SSs) % Hh, ow = (wsp + SSs) % Wd;  // undo roll(-3,-3)
      ro = bimg * Hh * Wd + oh * Wd + ow;
      int rr = (hs  < Hh - WSw) ? 0 : ((hs  < Hh - SSs) ? 1 : 2);
      int cc = (wsp < Wd - WSw) ? 0 : ((wsp < Wd - SSs) ? 1 : 2);
      id = rr * 3 + cc;
    }
    rowoff[t] = ro;
    rid[t]    = id;
  }
  __syncthreads();

  // zero-fill padded rows 49..63 (cols 0..191) of the three tiles
  for (int i = tid; i < 3 * 15 * 96; i += 192) {
    int t3 = i / (15 * 96);
    int j  = i % (15 * 96);
    int r  = 49 + j / 96, c = j % 96;
    ((uint32_t*)(qs + t3 * 64 * TSTR))[r * (TSTR / 2) + c] = 0;
  }
  // async-gather valid rows (16B chunks): Qn/Kn/Vb rows -> LDS tiles (ASYNCcnt path)
  {
    const __bf16* src[3] = {Qn, Kn, Vb};
#pragma unroll
    for (int t3 = 0; t3 < 3; ++t3) {
      __bf16* dst = qs + t3 * 64 * TSTR;
      for (int i = tid; i < Ntok * 24; i += 192) {
        int r = i / 24, ch = i % 24;  // 24 x 16B chunks per 192-elem row
        uint32_t la = (uint32_t)(uintptr_t)(dst + r * TSTR) + ch * 16;
        const char* ga = (const char*)(src[t3] + (long)rowoff[r] * Cc) + ch * 16;
        async_copy16(la, ga);
      }
    }
    wait_async0();
  }
  __syncthreads();

  const int wv = tid >> 5;  // head 0..5
  const int lane = tid & 31;
  const int mrow = lane & 15, hi = lane >> 4;
  const int cb = wv * HDm;

  // ---- S = Qn @ Kn^T  (4x4 tiles of 16x16, K=32) ----
  v16bf afr[4];
#pragma unroll
  for (int mt = 0; mt < 4; ++mt)
#pragma unroll
    for (int e = 0; e < 16; ++e)
      afr[mt][e] = qs[(mt * 16 + mrow) * TSTR + cb + a_kidx(e, hi)];
  v8f accS[4][4] = {};
#pragma unroll
  for (int nt = 0; nt < 4; ++nt) {
    v16bf bfr;
#pragma unroll
    for (int e = 0; e < 16; ++e)
      bfr[e] = ks[(nt * 16 + mrow) * TSTR + cb + b_kidx(e, hi)];
#pragma unroll
    for (int mt = 0; mt < 4; ++mt) accS[mt][nt] = wmma_bf16(afr[mt], bfr, accS[mt][nt]);
  }

  // ---- + rel-bias (LDS gather, pre-sigmoided) + shift mask, spill to LDS ----
  float*  Sw = Sb + wv * 64 * SSTR;
  __bf16* Pw = Pb + wv * 64 * PSTR;
#pragma unroll
  for (int mt = 0; mt < 4; ++mt)
#pragma unroll
    for (int nt = 0; nt < 4; ++nt)
#pragma unroll
      for (int r = 0; r < 8; ++r) {
        int m = mt * 16 + hi * 8 + r;
        int n = nt * 16 + mrow;
        float s = accS[mt][nt][r];
        if (n >= Ntok) {
          s = -1e30f;  // padded keys vanish in softmax
        } else if (m < Ntok) {
          int qy = m / WSw, qx = m % WSw, ky = n / WSw, kx = n % WSw;
          int bi = (qy - ky + WSw - 1) * (2 * WSw - 1) + (qx - kx + WSw - 1);
          s += biasS[bi * NHn + wv];
          if (rid[m] != rid[n]) s -= 100.f;
        }
        Sw[m * SSTR + n] = s;
      }
  __syncthreads();

  // ---- softmax (rows of this head's S) ----
  for (int m = lane; m < 64; m += 32) {
    float mx = -1e30f;
    for (int j = 0; j < 64; ++j) mx = fmaxf(mx, Sw[m * SSTR + j]);
    float sum = 0.f;
    for (int j = 0; j < 64; ++j) {
      float e = __expf(Sw[m * SSTR + j] - mx);
      Sw[m * SSTR + j] = e;
      sum += e;
    }
    float inv = 1.f / sum;
    for (int j = 0; j < 64; ++j) Pw[m * PSTR + j] = f2bf(Sw[m * SSTR + j] * inv);
  }
  __syncthreads();

  // ---- O = P @ V  (4x2 tiles, K=64 as 2 steps of 32) ----
  v8f accO[4][2] = {};
#pragma unroll
  for (int kk = 0; kk < 2; ++kk) {
    v16bf pa[4];
#pragma unroll
    for (int mt = 0; mt < 4; ++mt)
#pragma unroll
      for (int e = 0; e < 16; ++e)
        pa[mt][e] = Pw[(mt * 16 + mrow) * PSTR + kk * 32 + a_kidx(e, hi)];
#pragma unroll
    for (int nt = 0; nt < 2; ++nt) {
      v16bf vf;
#pragma unroll
      for (int e = 0; e < 16; ++e)
        vf[e] = vs[(kk * 32 + b_kidx(e, hi)) * TSTR + cb + nt * 16 + mrow];
#pragma unroll
      for (int mt = 0; mt < 4; ++mt) accO[mt][nt] = wmma_bf16(pa[mt], vf, accO[mt][nt]);
    }
  }
  // scatter back through inverse window/roll mapping -> natural (B,HW,C) layout
#pragma unroll
  for (int mt = 0; mt < 4; ++mt)
#pragma unroll
    for (int nt = 0; nt < 2; ++nt)
#pragma unroll
      for (int r = 0; r < 8; ++r) {
        int m = mt * 16 + hi * 8 + r;
        if (m < Ntok)
          O[(long)rowoff[m] * Cc + cb + nt * 16 + mrow] = accO[mt][nt][r];
      }
}

// ---------------- out = shortcut + LayerNorm(pin)*g + b  (wave per row) ----------------
__global__ __launch_bounds__(256) void ln_add_kernel(const float* __restrict__ shortcut,
                                                     const float* __restrict__ pin,
                                                     const float* __restrict__ g,
                                                     const float* __restrict__ b,
                                                     float* __restrict__ out) {
  int row  = blockIdx.x * 8 + (threadIdx.x >> 5);
  int lane = threadIdx.x & 31;
  const float* p = pin + (long)row * Cc;
  float v[6];
  float s = 0.f;
#pragma unroll
  for (int i = 0; i < 6; ++i) { v[i] = p[lane + i * 32]; s += v[i]; }
#pragma unroll
  for (int o = 16; o > 0; o >>= 1) s += __shfl_xor(s, o, 32);
  float mean = s * (1.f / Cc);
  float q = 0.f;
#pragma unroll
  for (int i = 0; i < 6; ++i) { float d = v[i] - mean; q += d * d; }
#pragma unroll
  for (int o = 16; o > 0; o >>= 1) q += __shfl_xor(q, o, 32);
  float rstd = rsqrtf(q * (1.f / Cc) + EPSc);
#pragma unroll
  for (int i = 0; i < 6; ++i) {
    int c = lane + i * 32;
    out[(long)row * Cc + c] = shortcut[(long)row * Cc + c] + (v[i] - mean) * rstd * g[c] + b[c];
  }
}

// ---------------- host-side orchestration ----------------
extern "C" void kernel_launch(void* const* d_in, const int* in_sizes, int n_in,
                              void* d_out, int out_size, void* d_ws, size_t ws_size,
                              hipStream_t stream) {
  const float* x           = (const float*)d_in[0];
  const float* ln1_g       = (const float*)d_in[1];
  const float* ln1_b       = (const float*)d_in[2];
  const float* ln2_g       = (const float*)d_in[3];
  const float* ln2_b       = (const float*)d_in[4];
  const float* q_w         = (const float*)d_in[5];
  const float* q_b         = (const float*)d_in[6];
  const float* k_w         = (const float*)d_in[7];
  const float* v_w         = (const float*)d_in[8];
  const float* v_b         = (const float*)d_in[9];
  const float* p_w         = (const float*)d_in[10];
  const float* p_b         = (const float*)d_in[11];
  const float* logit_scale = (const float*)d_in[12];
  const float* cpb_w0      = (const float*)d_in[13];
  const float* cpb_b0      = (const float*)d_in[14];
  const float* cpb_w1      = (const float*)d_in[15];
  const float* fc1_w       = (const float*)d_in[16];
  const float* fc1_b       = (const float*)d_in[17];
  const float* fc2_w       = (const float*)d_in[18];
  const float* fc2_b       = (const float*)d_in[19];

  char* ws = (char*)d_ws;
  const size_t SZH = (size_t)Mrows * Cc * 2;  // bf16 plane
  const size_t SZF = (size_t)Mrows * Cc * 4;  // f32 plane
  float*  bias_t = (float*)ws;                // 169*6 f32 (pre-sigmoided)
  float*  scales = (float*)(ws + 4056);       // 6 f32
  char*   base   = ws + 4096;
  __bf16* Qn   = (__bf16*)(base);
  __bf16* Kn   = (__bf16*)(base + SZH);
  __bf16* Vb   = (__bf16*)(base + 2 * SZH);
  float*  Obuf = (float*)(base + 3 * SZH);
  float*  Pbuf = (float*)(base + 3 * SZH + SZF);
  __bf16* FC1  = (__bf16*)(base);   // overlays Qn/Kn/Vb/Obuf (dead by then); 154.1 MB fits
  float*  MLP  = Pbuf;              // reuse (Pbuf dead after hidden)

  cpb_kernel<<<1, 256, 0, stream>>>(logit_scale, cpb_w0, cpb_b0, cpb_w1, bias_t, scales);

  dim3 blk(256);
  dim3 gC(Cc / 64, Mrows / 64);
  gemm_ws<float, EPI_BF16><<<gC, blk, 0, stream>>>(x, q_w, q_b, Qn, Cc, Cc);
  gemm_ws<float, EPI_BF16><<<gC, blk, 0, stream>>>(x, k_w, nullptr, Kn, Cc, Cc);
  gemm_ws<float, EPI_BF16><<<gC, blk, 0, stream>>>(x, v_w, v_b, Vb, Cc, Cc);

  norm_qk_kernel<<<(Mrows * NHn + 255) / 256, 256, 0, stream>>>(Qn, Kn, scales);

  attn_kernel<<<Bn * NWn, 192, ATTN_SMEM, stream>>>(Qn, Kn, Vb, bias_t, Obuf);

  gemm_ws<float, EPI_F32><<<gC, blk, 0, stream>>>(Obuf, p_w, p_b, Pbuf, Cc, Cc);

  ln_add_kernel<<<Mrows / 8, 256, 0, stream>>>(x, Pbuf, ln1_g, ln1_b, (float*)d_out);

  dim3 gFC1(4 * Cc / 64, Mrows / 64);
  gemm_ws<float, EPI_GELU_BF16><<<gFC1, blk, 0, stream>>>((const float*)d_out, fc1_w, fc1_b,
                                                          FC1, Cc, 4 * Cc);
  gemm_ws<__bf16, EPI_F32><<<gC, blk, 0, stream>>>(FC1, fc2_w, fc2_b, MLP, 4 * Cc, Cc);

  ln_add_kernel<<<Mrows / 8, 256, 0, stream>>>((const float*)d_out, MLP, ln2_g, ln2_b,
                                               (float*)d_out);
}
